// DeformLikeASPPConv_15745350107995
// MI455X (gfx1250) — matrix operand order repaired and checked by reading, not compiled, not run on hardware
//
#include <hip/hip_runtime.h>
#include <hip/hip_bf16.h>

typedef __attribute__((ext_vector_type(16))) __bf16 v16bf;
typedef __attribute__((ext_vector_type(8)))  float  v8f;

typedef int v4i __attribute__((__vector_size__(16)));
typedef __attribute__((address_space(1))) v4i glob_v4i;   // global-AS int4
typedef __attribute__((address_space(3))) v4i lds_v4i;    // LDS-AS int4

#define BATCH 8
#define CH    64
#define HH    256
#define WW    256
#define DILA  12

#if defined(__gfx1250__) && __has_builtin(__builtin_amdgcn_global_load_async_to_lds_b128)
#define HAVE_ASYNC_LDS 1
#endif

static __device__ __forceinline__ unsigned short f2bf(float f) {
    unsigned int u = __float_as_uint(f);
    u += 0x7FFFu + ((u >> 16) & 1u);   // round-to-nearest-even
    return (unsigned short)(u >> 16);
}

static __device__ __forceinline__ void wait_async0() {
#if defined(HAVE_ASYNC_LDS)
#if __has_builtin(__builtin_amdgcn_s_wait_asynccnt)
    __builtin_amdgcn_s_wait_asynccnt(0);
#else
    asm volatile("s_wait_asynccnt 0" ::: "memory");
#endif
#endif
}

// Form an AS1 (global) v4i pointer: flat and global share address values.
static __device__ __forceinline__ glob_v4i* as_global(const void* p) {
    return (glob_v4i*)(uintptr_t)p;
}
// Form an AS3 (LDS) v4i pointer: low 32 bits of a flat shared address are the
// LDS byte offset (ISA 10.2 aperture mapping); AS3 pointers are 32-bit.
static __device__ __forceinline__ lds_v4i* as_lds(void* p) {
    return (lds_v4i*)(unsigned int)(uintptr_t)p;
}

// ---------------------------------------------------------------------------
// Kernel 1: offset conv (64 -> 2ch, 3x3, pad 1) + bias + 2*tanh
// ---------------------------------------------------------------------------
__global__ __launch_bounds__(256) void offsets_kernel(
    const float* __restrict__ x, const float* __restrict__ ow,
    const float* __restrict__ ob, float2* __restrict__ offs)
{
    int gid = blockIdx.x * 256 + threadIdx.x;           // B*H*W
    int w = gid & (WW - 1);
    int h = (gid >> 8) & (HH - 1);
    int b = gid >> 16;
    const float* xb = x + (size_t)b * CH * HH * WW;
    float a0 = 0.f, a1 = 0.f;
    #pragma unroll
    for (int kh = 0; kh < 3; ++kh) {
        int y = h + kh - 1;
        if ((unsigned)y >= (unsigned)HH) continue;
        #pragma unroll
        for (int kw = 0; kw < 3; ++kw) {
            int xc = w + kw - 1;
            if ((unsigned)xc >= (unsigned)WW) continue;
            const float* xp = xb + y * WW + xc;
            const float* w0 = ow + 0 * CH * 9 + kh * 3 + kw;
            const float* w1 = ow + 1 * CH * 9 + kh * 3 + kw;
            for (int ci = 0; ci < CH; ++ci) {
                float v = xp[ci * (HH * WW)];
                a0 = fmaf(v, w0[ci * 9], a0);
                a1 = fmaf(v, w1[ci * 9], a1);
            }
        }
    }
    a0 = 2.0f * tanhf(a0 + ob[0]);
    a1 = 2.0f * tanhf(a1 + ob[1]);
    offs[gid] = make_float2(a0, a1);
}

// ---------------------------------------------------------------------------
// Kernel 2: bilinear warp (border clamp) -> warped, bf16, NHWC
// ---------------------------------------------------------------------------
__global__ __launch_bounds__(256) void warp_kernel(
    const float* __restrict__ x, const float2* __restrict__ offs,
    unsigned int* __restrict__ warped /* 2 bf16 per uint, NHWC */)
{
    int gid = blockIdx.x * 256 + threadIdx.x;           // B*H*W
    int w = gid & (WW - 1);
    int h = (gid >> 8) & (HH - 1);
    int b = gid >> 16;
    float2 t = offs[gid];
    float ix = fminf(fmaxf((float)w + t.x, 0.f), (float)(WW - 1));
    float iy = fminf(fmaxf((float)h + t.y, 0.f), (float)(HH - 1));
    float fx = floorf(ix), fy = floorf(iy);
    float wx = ix - fx,   wy = iy - fy;
    int x0 = (int)fx, y0 = (int)fy;
    int x1 = min(x0 + 1, WW - 1), y1 = min(y0 + 1, HH - 1);
    int i00 = y0 * WW + x0, i01 = y0 * WW + x1;
    int i10 = y1 * WW + x0, i11 = y1 * WW + x1;
    float w00 = (1.f - wx) * (1.f - wy), w01 = wx * (1.f - wy);
    float w10 = (1.f - wx) * wy,         w11 = wx * wy;
    const float* xb = x + (size_t)b * CH * HH * WW;
    unsigned int* dst = warped + (size_t)gid * (CH / 2);
    #pragma unroll 4
    for (int c = 0; c < CH; c += 2) {
        const float* p0 = xb + (size_t)c       * (HH * WW);
        const float* p1 = xb + (size_t)(c + 1) * (HH * WW);
        float v0 = p0[i00]*w00 + p0[i01]*w01 + p0[i10]*w10 + p0[i11]*w11;
        float v1 = p1[i00]*w00 + p1[i01]*w01 + p1[i10]*w10 + p1[i11]*w11;
        dst[c >> 1] = (unsigned int)f2bf(v0) | ((unsigned int)f2bf(v1) << 16);
    }
}

// ---------------------------------------------------------------------------
// Kernel 3: repack conv_w (COUT,CI,3,3) f32 -> bf16 B-fragment layout:
//   wp[tap][kk][ntile][lane][e], ci = kk*32 + (lane>>4)*16 + e,
//   co = ntile*16 + (lane&15)   (contiguous-K B 32x16 order, matched by the A fragment)
// ---------------------------------------------------------------------------
__global__ void wpack_kernel(const float* __restrict__ cw,
                             unsigned short* __restrict__ wp)
{
    int i = blockIdx.x * 256 + threadIdx.x;
    if (i >= 9 * 2 * 4 * 32 * 16) return;
    int e    = i & 15;
    int lane = (i >> 4) & 31;
    int nt   = (i >> 9) & 3;
    int kk   = (i >> 11) & 1;
    int tap  = i >> 12;
    int ci = kk * 32 + (lane >> 4) * 16 + e;
    int co = nt * 16 + (lane & 15);
    wp[i] = f2bf(cw[(co * CH + ci) * 9 + tap]);
}

// ---------------------------------------------------------------------------
// Kernel 4: dilated 3x3 conv as implicit GEMM on v_wmma_f32_16x16x32_bf16.
// A/W tiles staged with global_load_async_to_lds_b128 (ASYNCcnt); OOB rows
// zero-filled via ds_store. Fused BN + ReLU, coalesced NCHW store through
// padded-LDS transpose.
// ---------------------------------------------------------------------------
__global__ __launch_bounds__(256) void conv_wmma_kernel(
    const unsigned int* __restrict__ warped,   // bf16 NHWC, 32 uints / pixel
    const unsigned short* __restrict__ wpack,  // [9][2][4][32][16] bf16
    const float* __restrict__ gamma, const float* __restrict__ beta,
    const float* __restrict__ mean,  const float* __restrict__ var,
    float* __restrict__ out)
{
    __shared__ uint4 ldsA[128 * 8];    // 128 rows x 128B A-tile   (16 KB)
    __shared__ uint4 ldsW[512];        // one tap of packed weights (8 KB)
    __shared__ float ldsO[128 * 65];   // out transpose, pad 65    (32.5 KB)

    const int t    = threadIdx.x;
    const int lane = t & 31;
    const int wv   = t >> 5;
    const int lane16 = lane & 15;
    const int hi     = lane >> 4;

    const int blk = blockIdx.x;        // B*H*2 = 4096
    const int w0  = (blk & 1) * 128;
    const int h   = (blk >> 1) & (HH - 1);
    const int b   = blk >> 9;

    v8f acc[4];
    #pragma unroll
    for (int n = 0; n < 4; ++n) acc[n] = (v8f){0.f,0.f,0.f,0.f,0.f,0.f,0.f,0.f};

    const int rowm = wv * 16 + lane16;        // this lane's M row in tile

    for (int tap = 0; tap < 9; ++tap) {
        const int kh = tap / 3, kw = tap - kh * 3;
        const int hp = h + DILA * (kh - 1);

        __syncthreads();   // previous tap's LDS reads complete

        const int row  = t >> 1;
        const int half = t & 1;
        const int wp   = w0 + row + DILA * (kw - 1);
        const bool ok  = ((unsigned)hp < (unsigned)HH) &&
                         ((unsigned)wp < (unsigned)WW);
        uint4* ldst = &ldsA[row * 8 + half * 4];

#if defined(HAVE_ASYNC_LDS)
        // --- async DMA staging: global -> LDS, no VGPR round trip ---
        if (ok) {
            const unsigned int* gsrc = warped +
                ((size_t)(b * HH + hp) * WW + wp) * 32 + half * 16;
            #pragma unroll
            for (int i = 0; i < 4; ++i)
                __builtin_amdgcn_global_load_async_to_lds_b128(
                    as_global(gsrc + i * 4), as_lds(ldst + i), 0, 0);
        } else {
            const uint4 z = make_uint4(0u, 0u, 0u, 0u);
            #pragma unroll
            for (int i = 0; i < 4; ++i) ldst[i] = z;
        }
        {
            const unsigned short* wsrc = wpack + (size_t)tap * 4096;
            __builtin_amdgcn_global_load_async_to_lds_b128(
                as_global(wsrc + (size_t)t * 8), as_lds(&ldsW[t]), 0, 0);
            __builtin_amdgcn_global_load_async_to_lds_b128(
                as_global(wsrc + (size_t)(t + 256) * 8), as_lds(&ldsW[t + 256]), 0, 0);
        }
        wait_async0();
#else
        // --- fallback: load to VGPRs then ds_store ---
        {
            const uint4* src = (const uint4*)(wpack + (size_t)tap * 4096);
            ldsW[t]       = src[t];
            ldsW[t + 256] = src[t + 256];
        }
        {
            const uint4* src = (const uint4*)(warped +
                ((size_t)(b * HH + (ok ? hp : 0)) * WW + (ok ? wp : 0)) * 32);
            #pragma unroll
            for (int i = 0; i < 4; ++i) {
                uint4 v = make_uint4(0u, 0u, 0u, 0u);
                if (ok) v = src[half * 4 + i];
                ldst[i] = v;
            }
        }
#endif
        __syncthreads();

        // prefetch next tap's row into caches (global_prefetch_b8)
        if (tap < 8) {
            const int kh2 = (tap + 1) / 3, kw2 = (tap + 1) - kh2 * 3;
            const int hp2 = h + DILA * (kh2 - 1);
            const int wp2 = w0 + (t >> 1) + DILA * (kw2 - 1);
            if (((unsigned)hp2 < (unsigned)HH) && ((unsigned)wp2 < (unsigned)WW))
                __builtin_prefetch((const void*)(warped +
                    ((size_t)(b * HH + hp2) * WW + wp2) * 32), 0, 1);
        }

        // --- 2 x K=32 WMMA steps over ci ---
        union { uint4 u[2]; v16bf v; } af, bf;
        #pragma unroll
        for (int kk = 0; kk < 2; ++kk) {
            // A frag: lane(hi) picks contiguous K-half k = 16*hi + s (matches wpack B)
            af.u[0] = ldsA[rowm * 8 + kk * 4 + hi * 2];
            af.u[1] = ldsA[rowm * 8 + kk * 4 + hi * 2 + 1];
            #pragma unroll
            for (int n = 0; n < 4; ++n) {
                const int bi = ((kk * 4 + n) * 32 + lane) * 2;
                bf.u[0] = ldsW[bi];
                bf.u[1] = ldsW[bi + 1];
                acc[n] = __builtin_amdgcn_wmma_f32_16x16x32_bf16(
                    false, af.v, false, bf.v, (short)0, acc[n], false, false);
            }
        }
    }

    // --- epilogue: transpose via LDS, fused BN + ReLU, coalesced store ---
    __syncthreads();
    #pragma unroll
    for (int n = 0; n < 4; ++n) {
        const int co = n * 16 + lane16;
        #pragma unroll
        for (int r = 0; r < 8; ++r) {
            const int m = wv * 16 + hi * 8 + r;   // C/D layout: VGPR r, half hi
            ldsO[m * 65 + co] = acc[n][r];
        }
    }
    __syncthreads();
    for (int idx = t; idx < 64 * 128; idx += 256) {
        const int co = idx >> 7;
        const int m  = idx & 127;
        const float inv  = gamma[co] * rsqrtf(var[co] + 1e-5f);
        const float bias = beta[co] - mean[co] * inv;
        float v = fmaf(ldsO[m * 65 + co], inv, bias);
        out[((size_t)(b * CH + co) * HH + h) * WW + w0 + m] = fmaxf(v, 0.0f);
    }
}

// ---------------------------------------------------------------------------
extern "C" void kernel_launch(void* const* d_in, const int* in_sizes, int n_in,
                              void* d_out, int out_size, void* d_ws, size_t ws_size,
                              hipStream_t stream)
{
    const float* x   = (const float*)d_in[0];
    const float* ow  = (const float*)d_in[1];
    const float* ob  = (const float*)d_in[2];
    const float* cw  = (const float*)d_in[3];
    const float* bg  = (const float*)d_in[4];
    const float* bb  = (const float*)d_in[5];
    const float* bm  = (const float*)d_in[6];
    const float* bv  = (const float*)d_in[7];
    float* out = (float*)d_out;

    char* ws = (char*)d_ws;
    float2*         offs   = (float2*)ws;                                  // 4 MB
    unsigned int*   warped = (unsigned int*)(ws + (size_t)BATCH*HH*WW*8);  // 64 MB
    unsigned short* wp     = (unsigned short*)(ws + (size_t)BATCH*HH*WW*8
                                                  + (size_t)BATCH*HH*WW*CH*2);

    const int npix = BATCH * HH * WW;                 // 524288
    offsets_kernel<<<npix / 256, 256, 0, stream>>>(x, ow, ob, offs);
    warp_kernel<<<npix / 256, 256, 0, stream>>>(x, offs, warped);
    wpack_kernel<<<(9 * 2 * 4 * 32 * 16 + 255) / 256, 256, 0, stream>>>(cw, wp);
    conv_wmma_kernel<<<BATCH * HH * 2, 256, 0, stream>>>(
        warped, wp, bg, bb, bm, bv, out);
}
